// GATQN_79817672229315
// MI455X (gfx1250) — matrix operand order repaired and checked by reading, not compiled, not run on hardware
//
#include <hip/hip_runtime.h>

#define BB 4
#define NWK 256
#define NTK 256
#define DD 64
#define TT 3
#define N2 512

typedef __bf16  v16bf __attribute__((ext_vector_type(16)));
typedef float   v8f   __attribute__((ext_vector_type(8)));

__device__ __forceinline__ unsigned short f2bfbits(float f) {
  unsigned u = __builtin_bit_cast(unsigned, f);
  unsigned r = u + 0x7FFFu + ((u >> 16) & 1u);   // round-to-nearest-even
  return (unsigned short)(r >> 16);
}
__device__ __forceinline__ float lrelu(float x) { return x > 0.f ? x : 0.2f * x; }

__device__ __forceinline__ v8f wmma_bf16(v16bf a, v16bf b, v8f c) {
  return __builtin_amdgcn_wmma_f32_16x16x32_bf16(false, a, false, b, (short)0, c, false, false);
}

// Build a v16bf fragment from two contiguous 16-byte LDS runs (aligned).
__device__ __forceinline__ v16bf load_frag2(const unsigned short* p0,
                                            const unsigned short* p1) {
  union { v16bf v; uint4 q[2]; } u;
  u.q[0] = *(const uint4*)p0;
  u.q[1] = *(const uint4*)p1;
  return u.v;
}

// A fragment: 16x32 bf16 from row-major LDS tile sA[16][64], K-step ks.
// ISA 7.12.2: M = lane&15; lanes 0-15 take K kb..kb+7 / 16+kb.., kb=0;
// lanes 16-31 the same with kb=8.  Both runs are contiguous u16 spans.
__device__ __forceinline__ v16bf load_afrag(const unsigned short* sA, int lane, int ks) {
  int m  = lane & 15;
  int kb = (lane < 16) ? 0 : 8;
  const unsigned short* base = sA + m * 64 + ks * 32 + kb;
  return load_frag2(base, base + 16);
}
// B fragment from TRANSPOSED LDS weights sWt[n][K] (n-major, 64 K per row):
// N = nt*16 + (lane&15); lanes 0-15 hold K ks*32+0..7,16..23; lanes 16-31 K ks*32+8..15,24..31.
// Two contiguous 16B runs (same order as the A fragment).
__device__ __forceinline__ v16bf load_bfragT(const unsigned short* sWt, int lane,
                                             int ks, int nt) {
  int n  = nt * 16 + (lane & 15);
  int kb = (lane < 16) ? 0 : 8;
  const unsigned short* base = sWt + n * 64 + ks * 32 + kb;
  return load_frag2(base, base + 16);
}

// ---------------------------------------------------------------- prep kernels
__global__ void k_prep_wna(const float* __restrict__ wloc, const float* __restrict__ cap,
                           const float* __restrict__ ct, const float* __restrict__ rev,
                           const float* __restrict__ tpd, const float* __restrict__ unst,
                           float* __restrict__ wna) {
  int idx = blockIdx.x * blockDim.x + threadIdx.x;
  if (idx >= BB * NWK) return;
  float v[7];
  v[0] = wloc[idx * 2]; v[1] = wloc[idx * 2 + 1];
  v[2] = cap[idx]; v[3] = ct[idx]; v[4] = rev[idx]; v[5] = tpd[idx]; v[6] = unst[idx];
  float n = 0.f;
  for (int i = 0; i < 6; ++i) n += v[i] * v[i];
  n = fmaxf(sqrtf(n), 1e-12f);
  for (int i = 0; i < 6; ++i) v[i] /= n;
  for (int i = 0; i < 7; ++i) wna[idx * 7 + i] = v[i];
}

__global__ void k_prep_tna(const float* __restrict__ tloc, const float* __restrict__ cw,
                           const float* __restrict__ bud, const float* __restrict__ tqr,
                           const float* __restrict__ nwpt, const float* __restrict__ inc,
                           const float* __restrict__ cpp, float* __restrict__ tna) {
  int idx = blockIdx.x * blockDim.x + threadIdx.x;
  if (idx >= BB * NTK) return;
  float v[8];
  v[0] = tloc[idx * 2]; v[1] = tloc[idx * 2 + 1];
  v[2] = cw[idx]; v[3] = bud[idx]; v[4] = tqr[idx]; v[5] = nwpt[idx];
  v[6] = inc[idx]; v[7] = cpp[idx];
  float n = 0.f;
  for (int i = 0; i < 7; ++i) n += v[i] * v[i];
  n = fmaxf(sqrtf(n), 1e-12f);
  for (int i = 0; i < 7; ++i) v[i] /= n;
  for (int i = 0; i < 8; ++i) tna[idx * 8 + i] = v[i];
}

__global__ void k_prep_wea(const float* __restrict__ wd, const float* __restrict__ jc,
                           float* __restrict__ wea) {
  int idx = blockIdx.x * blockDim.x + threadIdx.x;
  if (idx >= BB * NWK * NWK) return;
  float a = wd[idx], b = jc[idx];
  float n = fmaxf(sqrtf(a * a + b * b), 1e-12f);
  wea[idx * 2] = a / n; wea[idx * 2 + 1] = b / n;
}

__global__ void k_prep_tea(const float* __restrict__ td, float* __restrict__ tea) {
  int idx = blockIdx.x * blockDim.x + threadIdx.x;
  if (idx >= BB * NTK * NTK) return;
  float a = td[idx];
  tea[idx] = a / fmaxf(fabsf(a), 1e-12f);
}

__global__ void k_prep_wta(const float* __restrict__ adj, const float* __restrict__ act,
                           float* __restrict__ wta) {
  int idx = blockIdx.x * blockDim.x + threadIdx.x;
  if (idx >= BB * NWK * NTK) return;
  wta[idx] = adj[idx] * (1.f - act[idx]);
}

__global__ void k_prep_wtea(const float* __restrict__ wtd, const float* __restrict__ wqr,
                            const int* __restrict__ paths, float* __restrict__ wtea) {
  int idx = blockIdx.x * blockDim.x + threadIdx.x;  // b*NW + i
  if (idx >= BB * NWK) return;
  const int* prow = paths + idx * NTK;
  int amax = 0, best = prow[0], any = (best >= 0);
  for (int j = 1; j < NTK; ++j) {
    int pv = prow[j];
    if (pv > best) { best = pv; amax = j; }
    any |= (pv >= 0);
  }
  for (int j = 0; j < NTK; ++j) {
    float r0 = wtd[idx * NTK + j];
    float r1 = wqr[idx * NTK + j];
    float r2 = (prow[j] < 0) ? 1.f : 0.f;
    float r3 = (any && j == amax) ? 1.f : 0.f;
    float n = fmaxf(sqrtf(r0 * r0 + r1 * r1 + r2 * r2), 1e-12f);
    float* o = wtea + (idx * NTK + j) * 4;
    o[0] = r0 / n; o[1] = r1 / n; o[2] = r2 / n; o[3] = r3;
  }
}

// ------------------------------------------------ small linear: [rows,fi]->[rows,64], lrelu
__global__ void __launch_bounds__(32) k_lin_act(const float* __restrict__ in,
                                                const float* __restrict__ W,
                                                const float* __restrict__ bias,
                                                float* __restrict__ out, int rows, int fi) {
  int row = blockIdx.x, lane = threadIdx.x;
  if (row >= rows) return;
  float iv[8];
  for (int r = 0; r < fi; ++r) iv[r] = in[row * fi + r];
#pragma unroll
  for (int t = 0; t < 2; ++t) {
    int d = lane + t * 32;
    float acc = bias[d];
    for (int r = 0; r < fi; ++r) acc += iv[r] * W[r * 64 + d];
    out[row * 64 + d] = lrelu(acc);
  }
}

// ------------------------------------------------ WMMA GEMM: C[rows,64] = A[rows,64] @ W[64,64]
__global__ void __launch_bounds__(32) k_gemm64_bf16(const float* __restrict__ A,
                                                    const float* __restrict__ W,
                                                    float* __restrict__ C, int rows) {
  __shared__ __align__(16) unsigned short sA[16 * 64];
  __shared__ __align__(16) unsigned short sWt[64 * 64];  // transposed: sWt[n][K]
  int lane = threadIdx.x;
  int r0 = blockIdx.x * 16;
  if (r0 >= rows) return;
  for (int i = lane; i < 64 * 64; i += 32) {
    int r = i >> 6, c = i & 63;
    sWt[c * 64 + r] = f2bfbits(W[i]);
  }
  for (int i = lane; i < 16 * 64; i += 32) {
    int rr = i >> 6, cc = i & 63;
    sA[i] = f2bfbits(A[(r0 + rr) * 64 + cc]);
  }
  __syncthreads();
  v16bf bf[2][4];
#pragma unroll
  for (int ks = 0; ks < 2; ++ks)
#pragma unroll
    for (int nt = 0; nt < 4; ++nt) bf[ks][nt] = load_bfragT(sWt, lane, ks, nt);
  v16bf a0 = load_afrag(sA, lane, 0);
  v16bf a1 = load_afrag(sA, lane, 1);
  v8f z = {0.f, 0.f, 0.f, 0.f, 0.f, 0.f, 0.f, 0.f};
  v8f acc[4] = {z, z, z, z};
#pragma unroll
  for (int nt = 0; nt < 4; ++nt) {
    acc[nt] = wmma_bf16(a0, bf[0][nt], acc[nt]);
    acc[nt] = wmma_bf16(a1, bf[1][nt], acc[nt]);
  }
  int m0 = (lane < 16) ? 0 : 8;
  int n = lane & 15;
#pragma unroll
  for (int nt = 0; nt < 4; ++nt)
#pragma unroll
    for (int rv = 0; rv < 8; ++rv)
      C[(r0 + rv + m0) * 64 + nt * 16 + n] = acc[nt][rv];
}

// ------------------------------------------------ fused GAT layer: one wave per (b, i)
__global__ void __launch_bounds__(32) k_gat(
    const float* __restrict__ qbuf, const float* __restrict__ kbuf,
    const float* __restrict__ sbuf,
    const float* __restrict__ eraw, int edim, int etrans,
    const float* __restrict__ eW, const float* __restrict__ eB,
    const float* __restrict__ We, const float* __restrict__ vvec,
    const float* __restrict__ adj, int atrans,
    float* __restrict__ hout, int Nq, int Nk) {
  __shared__ __align__(16) unsigned short sWeT[64 * 64];  // transposed: sWeT[d][de]
  __shared__ __align__(16) unsigned short sE[16 * 64];
  __shared__ float sEW[4 * 64];
  __shared__ float sEB[64];

  int lane = threadIdx.x;
  int gi = blockIdx.x;
  int b = gi / Nq, i = gi % Nq;

  for (int t = lane; t < 64 * 64; t += 32) {
    int r = t >> 6, c = t & 63;
    sWeT[c * 64 + r] = f2bfbits(We[t]);
  }
  for (int t = lane; t < edim * 64; t += 32) sEW[t] = eW[t];
  for (int t = lane; t < 64; t += 32) sEB[t] = eB[t];
  __syncthreads();

  v16bf wfrag[2][4];
#pragma unroll
  for (int ks = 0; ks < 2; ++ks)
#pragma unroll
    for (int nt = 0; nt < 4; ++nt) wfrag[ks][nt] = load_bfragT(sWeT, lane, ks, nt);

  int nsel = lane & 15;
  int m0 = (lane < 16) ? 0 : 8;
  float qv[4], vv[4];
#pragma unroll
  for (int nt = 0; nt < 4; ++nt) {
    qv[nt] = qbuf[(b * Nq + i) * 64 + nt * 16 + nsel];
    vv[nt] = vvec[nt * 16 + nsel];
  }

  float runM = -3.0e38f, runS = 0.f;
  float macc[4] = {0.f, 0.f, 0.f, 0.f};

  int jr = lane >> 1;
  int de0 = (lane & 1) * 32;

  for (int j0 = 0; j0 < Nk; j0 += 16) {
    __syncthreads();  // protect sE from previous tile's readers
    {   // compute bf16 edge-embedding tile: lane -> row (lane>>1), 32 cols
      int j = j0 + jr;
      float raw[4];
      int eidx = etrans ? ((b * Nk + j) * Nq + i) : ((b * Nq + i) * Nk + j);
      for (int r = 0; r < edim; ++r) raw[r] = eraw[eidx * edim + r];
      // prefetch the next tile's raw-edge row (global_prefetch_b8)
      if (j0 + 16 < Nk) {
        int jn = j + 16;
        int enx = etrans ? ((b * Nk + jn) * Nq + i) : ((b * Nq + i) * Nk + jn);
        __builtin_prefetch(&eraw[enx * edim], 0, 1);
      }
      for (int dd = 0; dd < 32; ++dd) {
        int de = de0 + dd;
        float a = sEB[de];
        for (int r = 0; r < edim; ++r) a += raw[r] * sEW[r * 64 + de];
        sE[jr * 64 + de] = f2bfbits(lrelu(a));
      }
    }
    __syncthreads();
    v16bf a0 = load_afrag(sE, lane, 0);
    v16bf a1 = load_afrag(sE, lane, 1);
    v8f z = {0.f, 0.f, 0.f, 0.f, 0.f, 0.f, 0.f, 0.f};
    v8f m4[4] = {z, z, z, z};
#pragma unroll
    for (int nt = 0; nt < 4; ++nt) {
      m4[nt] = wmma_bf16(a0, wfrag[0][nt], m4[nt]);
      m4[nt] = wmma_bf16(a1, wfrag[1][nt], m4[nt]);
    }
    // m = eW + k[j];  partial logits over this lane's 4 d-values
    float lg8[8];
#pragma unroll
    for (int rv = 0; rv < 8; ++rv) lg8[rv] = 0.f;
#pragma unroll
    for (int nt = 0; nt < 4; ++nt)
#pragma unroll
      for (int rv = 0; rv < 8; ++rv) {
        float kv = kbuf[(b * Nk + j0 + rv + m0) * 64 + nt * 16 + nsel];
        float m = m4[nt][rv] + kv;
        m4[nt][rv] = m;
        lg8[rv] += vv[nt] * lrelu(qv[nt] + m);
      }
    // reduce logits across the 16-lane half (d dimension)
#pragma unroll
    for (int sh = 1; sh < 16; sh <<= 1)
#pragma unroll
      for (int rv = 0; rv < 8; ++rv) lg8[rv] += __shfl_xor(lg8[rv], sh, 32);
    // mask + online softmax update
    float mk[8], cmax = -3.0e38f;
#pragma unroll
    for (int rv = 0; rv < 8; ++rv) {
      int j = j0 + rv + m0;
      float a = atrans ? adj[(b * Nk + j) * Nq + i] : adj[(b * Nq + i) * Nk + j];
      mk[rv] = (a > 0.f) ? 1.f : 0.f;
      lg8[rv] = (a > 0.f) ? lg8[rv] : -1e9f;
      cmax = fmaxf(cmax, lg8[rv]);
    }
    cmax = fmaxf(cmax, __shfl_xor(cmax, 16, 32));
    float newM = fmaxf(runM, cmax);
    float scale = __expf(runM - newM);
    runS *= scale;
#pragma unroll
    for (int nt = 0; nt < 4; ++nt) macc[nt] *= scale;
#pragma unroll
    for (int rv = 0; rv < 8; ++rv) {
      float w = __expf(lg8[rv] - newM);
      runS += w;                       // denominator includes masked ~0 terms
      float wm = w * mk[rv];           // alpha * mask for numerator
#pragma unroll
      for (int nt = 0; nt < 4; ++nt) macc[nt] += wm * m4[nt][rv];
    }
    runM = newM;
  }
  // combine lane halves (each accumulated its own 8 j rows per tile)
  runS += __shfl_xor(runS, 16, 32);
#pragma unroll
  for (int nt = 0; nt < 4; ++nt) macc[nt] += __shfl_xor(macc[nt], 16, 32);
  float inv = 1.f / runS;
  if (lane < 16) {
#pragma unroll
    for (int nt = 0; nt < 4; ++nt) {
      int d = nt * 16 + lane;
      float s = sbuf[(b * Nq + i) * 64 + d];
      hout[(b * Nq + i) * 64 + d] = lrelu(s + macc[nt] * inv);
    }
  }
}

// ------------------------------------------------ final cat-linear: act([i|h] @ W + b)
__global__ void __launch_bounds__(32) k_lin_cat(const float* __restrict__ ia,
                                                const float* __restrict__ hb,
                                                const float* __restrict__ W,
                                                const float* __restrict__ bias,
                                                float* __restrict__ out, int rows) {
  int row = blockIdx.x, lane = threadIdx.x;
  if (row >= rows) return;
#pragma unroll
  for (int t = 0; t < 2; ++t) {
    int d = lane + t * 32;
    float acc = bias[d];
    for (int k = 0; k < 64; ++k) acc += ia[row * 64 + k] * W[k * 64 + d];
    for (int k = 0; k < 64; ++k) acc += hb[row * 64 + k] * W[(64 + k) * 64 + d];
    out[row * 64 + d] = lrelu(acc);
  }
}

// ------------------------------------------------ attention pooling scores s[b,k]
__global__ void k_attn_score(const float* __restrict__ fhw, const float* __restrict__ fht,
                             const float* __restrict__ fhwb, const float* __restrict__ fhtb,
                             const float* __restrict__ Wat, const float* __restrict__ bat,
                             const float* __restrict__ wmlp, float* __restrict__ sraw) {
  __shared__ float sW[64 * 64];
  __shared__ float sb[64];
  __shared__ float sm[64];
  __shared__ float red[256];
  int b = blockIdx.x / 3, k = blockIdx.x % 3;
  int tid = threadIdx.x;
  for (int t = tid; t < 64 * 64; t += 256) sW[t] = Wat[t];
  for (int t = tid; t < 64; t += 256) { sb[t] = bat[t]; sm[t] = wmlp[t]; }
  __syncthreads();
  float part = 0.f;
  for (int n = tid; n < N2; n += 256) {
    const float* src = nullptr;
    if (k == 0)      { if (n < NWK)  src = &fhw[(b * NWK + n) * 64]; }
    else if (k == 1) { if (n >= NWK) src = &fht[(b * NTK + (n - NWK)) * 64]; }
    else             { src = (n < NWK) ? &fhwb[(b * NWK + n) * 64]
                                       : &fhtb[(b * NTK + (n - NWK)) * 64]; }
    float acc = 0.f;
    for (int d = 0; d < 64; ++d) {
      float ad = sb[d];
      if (src)
        for (int e = 0; e < 64; ++e) ad += src[e] * sW[e * 64 + d];
      acc += tanhf(ad) * sm[d];
    }
    part += acc;
  }
  red[tid] = part;
  __syncthreads();
  for (int s = 128; s > 0; s >>= 1) {
    if (tid < s) red[tid] += red[tid + s];
    __syncthreads();
  }
  if (tid == 0) sraw[b * 3 + k] = __expf(red[0] / (float)N2);
}

// ------------------------------------------------ final weighted mix -> out [B,512,64]
__global__ void k_mix(const float* __restrict__ fhw, const float* __restrict__ fht,
                      const float* __restrict__ fhwb, const float* __restrict__ fhtb,
                      const float* __restrict__ sraw, float* __restrict__ out) {
  int idx = blockIdx.x * blockDim.x + threadIdx.x;
  if (idx >= BB * N2 * DD) return;
  int b = idx / (N2 * DD);
  int rem = idx % (N2 * DD);
  int n = rem / DD, d = rem % DD;
  float s0 = sraw[b * 3 + 0], s1 = sraw[b * 3 + 1], s2 = sraw[b * 3 + 2];
  float tot = fmaxf(s0 + s1 + s2, 1e-8f);
  float v0 = (n < NWK) ? fhw[(b * NWK + n) * 64 + d] : 0.f;
  float v1 = (n < NWK) ? 0.f : fht[(b * NTK + (n - NWK)) * 64 + d];
  float v2 = (n < NWK) ? fhwb[(b * NWK + n) * 64 + d] : fhtb[(b * NTK + (n - NWK)) * 64 + d];
  out[idx] = (s0 * v0 + s1 * v1 + s2 * v2) / tot;
}

// ================================================================ host side
extern "C" void kernel_launch(void* const* d_in, const int* in_sizes, int n_in,
                              void* d_out, int out_size, void* d_ws, size_t ws_size,
                              hipStream_t stream) {
  (void)in_sizes; (void)n_in; (void)out_size; (void)ws_size;
  const float* worker_location = (const float*)d_in[1];
  const float* task_location   = (const float*)d_in[2];
  const float* worker_adj      = (const float*)d_in[3];
  const float* task_adj        = (const float*)d_in[4];
  const float* wt_adj          = (const float*)d_in[5];
  const float* wdist           = (const float*)d_in[7];
  const float* tdist           = (const float*)d_in[8];
  const float* wtdist          = (const float*)d_in[9];
  const float* unstopped       = (const float*)d_in[10];
  const float* incomplete      = (const float*)d_in[11];
  const float* rest_cap        = (const float*)d_in[12];
  const float* cur_time        = (const float*)d_in[13];
  const float* revenue         = (const float*)d_in[14];
  const float* tpd             = (const float*)d_in[15];
  const float* comp_win        = (const float*)d_in[16];
  const float* budgets         = (const float*)d_in[17];
  const float* nwpt            = (const float*)d_in[18];
  const float* tqr             = (const float*)d_in[19];
  const float* wqr             = (const float*)d_in[20];
  const float* jacc            = (const float*)d_in[21];
  const int*   paths           = (const int*)d_in[22];
  const float* action          = (const float*)d_in[23];
  const float* cpp             = (const float*)d_in[24];
  const int PB = 25;
  auto P = [&](int k) { return (const float*)d_in[PB + k]; };

  // workspace layout (floats)
  float* w = (float*)d_ws;
  float* WNAp = w;  w += BB * NWK * 7;
  float* TNAp = w;  w += BB * NTK * 8;
  float* WEAp = w;  w += BB * NWK * NWK * 2;
  float* TEAp = w;  w += BB * NTK * NTK;
  float* WTEAp = w; w += BB * NWK * NTK * 4;
  float* WTAp = w;  w += BB * NWK * NTK;
  const int HS = BB * 256 * 64;
  float* HB[8];
  for (int t = 0; t < 8; ++t) { HB[t] = w; w += HS; }
  float* IW = w;  w += HS;
  float* IT = w;  w += HS;
  float* IWB = w; w += HS;
  float* ITB = w; w += HS;
  float* QB = w;  w += HS;
  float* KB = w;  w += HS;
  float* SB = w;  w += HS;
  float* FHW = w;  w += HS;
  float* FHT = w;  w += HS;
  float* FHWB = w; w += HS;
  float* FHTB = w; w += HS;
  float* SRAW = w; w += BB * 3;

  // ---- prep
  k_prep_wna<<<(BB * NWK + 255) / 256, 256, 0, stream>>>(worker_location, rest_cap, cur_time,
                                                         revenue, tpd, unstopped, WNAp);
  k_prep_tna<<<(BB * NTK + 255) / 256, 256, 0, stream>>>(task_location, comp_win, budgets, tqr,
                                                         nwpt, incomplete, cpp, TNAp);
  k_prep_wea<<<(BB * NWK * NWK + 255) / 256, 256, 0, stream>>>(wdist, jacc, WEAp);
  k_prep_tea<<<(BB * NTK * NTK + 255) / 256, 256, 0, stream>>>(tdist, TEAp);
  k_prep_wta<<<(BB * NWK * NTK + 255) / 256, 256, 0, stream>>>(wt_adj, action, WTAp);
  k_prep_wtea<<<(BB * NWK + 255) / 256, 256, 0, stream>>>(wtdist, wqr, paths, WTEAp);

  // ---- initial embeds
  float *hw_c = HB[0], *hw_n = HB[1], *ht_c = HB[2], *ht_n = HB[3];
  float *hwb_c = HB[4], *hwb_n = HB[5], *htb_c = HB[6], *htb_n = HB[7];
  k_lin_act<<<BB * NWK, 32, 0, stream>>>(WNAp, P(0), P(1), hw_c, BB * NWK, 7);
  k_lin_act<<<BB * NTK, 32, 0, stream>>>(TNAp, P(4), P(5), ht_c, BB * NTK, 8);
  k_lin_act<<<BB * NWK, 32, 0, stream>>>(WNAp, P(8), P(9), hwb_c, BB * NWK, 7);
  k_lin_act<<<BB * NTK, 32, 0, stream>>>(TNAp, P(10), P(11), htb_c, BB * NTK, 8);
  k_lin_act<<<BB * NWK, 32, 0, stream>>>(WNAp, P(14), P(15), IW, BB * NWK, 7);
  k_lin_act<<<BB * NTK, 32, 0, stream>>>(TNAp, P(16), P(17), IT, BB * NTK, 8);
  k_lin_act<<<BB * NWK, 32, 0, stream>>>(WNAp, P(18), P(19), IWB, BB * NWK, 7);
  k_lin_act<<<BB * NTK, 32, 0, stream>>>(TNAp, P(20), P(21), ITB, BB * NTK, 8);

  auto run_gat = [&](const float* hq, const float* hk, const float* Wq, const float* Wk,
                     const float* Ws, const float* We_, const float* vv, const float* eraw,
                     int edim, const float* eW, const float* eB, int etrans, const float* adj,
                     int atrans, int Nq, int Nk, float* hout) {
    k_gemm64_bf16<<<(BB * Nq) / 16, 32, 0, stream>>>(hq, Wq, QB, BB * Nq);
    k_gemm64_bf16<<<(BB * Nk) / 16, 32, 0, stream>>>(hk, Wk, KB, BB * Nk);
    k_gemm64_bf16<<<(BB * Nq) / 16, 32, 0, stream>>>(hq, Ws, SB, BB * Nq);
    k_gat<<<BB * Nq, 32, 0, stream>>>(QB, KB, SB, eraw, edim, etrans, eW, eB, We_, vv, adj,
                                      atrans, hout, Nq, Nk);
  };

  const int M64 = 64 * 64;
  for (int it = 0; it < TT; ++it) {
    // worker self-GAT
    run_gat(hw_c, hw_c, P(33) + it * M64, P(34) + it * M64, P(36) + it * M64,
            P(35) + it * M64, P(37) + it * 64, WEAp, 2, P(2), P(3), 0, worker_adj, 0,
            NWK, NWK, hw_n);
    // task self-GAT
    run_gat(ht_c, ht_c, P(38) + it * M64, P(39) + it * M64, P(41) + it * M64,
            P(40) + it * M64, P(42) + it * 64, TEAp, 1, P(6), P(7), 0, task_adj, 0,
            NTK, NTK, ht_n);
    // bipartite worker<-task
    run_gat(hwb_c, htb_c, P(43) + it * M64, P(44) + it * M64, P(46) + it * M64,
            P(45) + it * M64, P(47) + it * 64, WTEAp, 4, P(12), P(13), 0, WTAp, 0,
            NWK, NTK, hwb_n);
    // bipartite task<-worker (transposed edge/adj)
    run_gat(htb_c, hwb_c, P(48) + it * M64, P(49) + it * M64, P(51) + it * M64,
            P(50) + it * M64, P(52) + it * 64, WTEAp, 4, P(12), P(13), 1, WTAp, 1,
            NTK, NWK, htb_n);
    float* t;
    t = hw_c;  hw_c = hw_n;   hw_n = t;
    t = ht_c;  ht_c = ht_n;   ht_n = t;
    t = hwb_c; hwb_c = hwb_n; hwb_n = t;
    t = htb_c; htb_c = htb_n; htb_n = t;
  }

  // ---- final aggregation
  k_lin_cat<<<BB * NWK, 32, 0, stream>>>(IW, hw_c, P(22), P(23), FHW, BB * NWK);
  k_lin_cat<<<BB * NTK, 32, 0, stream>>>(IT, ht_c, P(24), P(25), FHT, BB * NTK);
  k_lin_cat<<<BB * NWK, 32, 0, stream>>>(IWB, hwb_c, P(26), P(27), FHWB, BB * NWK);
  k_lin_cat<<<BB * NTK, 32, 0, stream>>>(ITB, htb_c, P(28), P(29), FHTB, BB * NTK);
  k_attn_score<<<BB * 3, 256, 0, stream>>>(FHW, FHT, FHWB, FHTB, P(30), P(31), P(32), SRAW);
  k_mix<<<(BB * N2 * DD + 255) / 256, 256, 0, stream>>>(FHW, FHT, FHWB, FHTB, SRAW,
                                                        (float*)d_out);
}
